// LightConeAttention_42202348650532
// MI455X (gfx1250) — matrix operand order repaired and checked by reading, not compiled, not run on hardware
//
#include <hip/hip_runtime.h>

// ---------------------------------------------------------------------------
// Light-cone attention for MI455X (gfx1250, wave32, WMMA).
// B=2, T=6, H=W=16, C=256, 8 heads x head_dim 32, N = 1536.
//
// Stage 0: prep      — x -> f16, Wqkv/Wo -> transposed f16 (one-off, ~1M elems)
// Stage 1: QKV GEMM  — 16x32 tile per wave, v_wmma_f32_16x16x32_f16
// Stage 2: attention — flash-style online softmax, light-cone tile skipping
// Stage 3: out proj  — 16x32 tile per wave, bias add, f32 output
//
// Every matmul operand is a per-lane contiguous 16/32-byte load in a
// contiguous k order (k = 16*(lane/16) + s) shared by A and B; all accumulation is f32.
// ---------------------------------------------------------------------------

#define Bb 2
#define Tt 6
#define Hh 16
#define WW 16
#define Cc 256
#define NH 8
#define HD 32
#define Nn (Tt * Hh * WW)   // 1536
#define BN (Bb * Nn)        // 3072
#define QT (Nn / 16)        // 96 query tiles per batch-image

#define XN (BN * Cc)        // 786432  x elements
#define WQN (Cc * 3 * Cc)   // 196608  Wqkv elements
#define WON (Cc * Cc)       // 65536   Wo elements

typedef __attribute__((ext_vector_type(16))) _Float16 v16h;
typedef __attribute__((ext_vector_type(8)))  _Float16 v8h;
typedef __attribute__((ext_vector_type(8)))  float    v8f;

union V16U { v16h v; v8h h[2]; };

__device__ __forceinline__ int iabs(int v) { return v < 0 ? -v : v; }

__device__ __forceinline__ v8f wmma_f16(v16h a, v16h b, v8f c) {
  // D = A(16x32 f16) x B(32x16 f16) + C(16x16 f32)
  return __builtin_amdgcn_wmma_f32_16x16x32_f16(
      /*neg_a=*/false, a, /*neg_b=*/false, b,
      /*c_mod=*/(short)0, c, /*reuse_a=*/false, /*reuse_b=*/false);
}

// ---- WMMA operand gathers (contiguous-k order, same for A and B) ----------------------------
// A 16x32 f16: lane L holds row M=L&15; K runs [kb,kb+15],
// kb = (L>=16)?16:0  -> one contiguous 32B load per lane.
__device__ __forceinline__ v16h loadA_f16(const _Float16* base, int ld) {
  int lane = threadIdx.x & 31;
  int m   = lane & 15;
  int kb  = (lane >> 4) << 4;           // 0 or 16
  const _Float16* p = base + m * ld;
  return *(const v16h*)(p + kb);
}

// B 32x16 f16 where element (k,n) lives at base[n*ld + k] (16 "columns" of the
// operand are contiguous in k) -> one 32B vector load per lane.
// Lane L: N = L&15, K runs [kb, kb+15], kb = (L>=16)?16:0.
__device__ __forceinline__ v16h loadBt_f16(const _Float16* base, int ld) {
  int lane = threadIdx.x & 31;
  int n  = lane & 15;
  int kb = (lane >> 4) << 4;            // 0 or 16
  return *(const v16h*)(base + n * ld + kb);
}

// ---------------------------------------------------------------------------
// Stage 0: one-off precision/layout prep.
//   xh[i]        = f16(x[i])                     (same layout, [BN][256])
//   WqkvT[c][k]  = f16(Wqkv[k][c])               ([768][256])
//   WoT[c][k]    = f16(Wo[k][c])                 ([256][256])
// Flat 1M work-items, reads coalesced.
// ---------------------------------------------------------------------------
__global__ __launch_bounds__(256) void lc_prep_kernel(
    const float* __restrict__ x, const float* __restrict__ Wqkv,
    const float* __restrict__ Wo, _Float16* __restrict__ xh,
    _Float16* __restrict__ WqkvT, _Float16* __restrict__ WoT) {
  int i = blockIdx.x * 256 + threadIdx.x;     // 0 .. 1048575 exactly
  if (i < XN) {
    xh[i] = (_Float16)x[i];
  } else if (i < XN + WQN) {
    int j = i - XN;
    int k = j / (3 * Cc);
    int c = j - k * (3 * Cc);
    WqkvT[c * Cc + k] = (_Float16)Wqkv[j];
  } else {
    int j = i - XN - WQN;
    int k = j >> 8;
    int c = j & 255;
    WoT[c * Cc + k] = (_Float16)Wo[j];
  }
}

// ---------------------------------------------------------------------------
// Stage 1: qkv = x @ Wqkv.  One wave per 16x32 output tile (A reused across
// two B tiles).  Q,K stored f16 row-major [BN][256]; V stored transposed
// [B][head][32][N] so the attention PV B-operand is contiguous.
// ---------------------------------------------------------------------------
__device__ __forceinline__ void store_qkv_half(
    v8f acc, int row0, int col0, int n, int mAdd,
    _Float16* __restrict__ qbuf, _Float16* __restrict__ kbuf,
    _Float16* __restrict__ vT) {
  int which = col0 >> 8;                      // 0:Q 1:K 2:V (no straddle)
  int cbase = col0 & 255;
#pragma unroll
  for (int e = 0; e < 8; ++e) {
    int m   = e + mAdd;
    int row = row0 + m;
    int col = cbase + n;
    _Float16 h = (_Float16)acc[e];
    if (which == 0) {
      qbuf[row * Cc + col] = h;
    } else if (which == 1) {
      kbuf[row * Cc + col] = h;
    } else {
      int bb = row / Nn;
      int nn = row - bb * Nn;
      int head = col >> 5;
      int d    = col & 31;
      vT[((bb * NH + head) * HD + d) * Nn + nn] = h;
    }
  }
}

__global__ __launch_bounds__(128) void lc_qkv_kernel(
    const _Float16* __restrict__ xh, const _Float16* __restrict__ WqkvT,
    _Float16* __restrict__ qbuf, _Float16* __restrict__ kbuf,
    _Float16* __restrict__ vT) {
  int wave = threadIdx.x >> 5;
  int tile = blockIdx.x * 4 + wave;           // 192 * 24 = 4608 tiles
  int rowT = tile / 24;
  int colP = tile % 24;
  int row0 = rowT * 16;
  int col0 = colP * 32;

  v8f acc0 = {0.f, 0.f, 0.f, 0.f, 0.f, 0.f, 0.f, 0.f};
  v8f acc1 = {0.f, 0.f, 0.f, 0.f, 0.f, 0.f, 0.f, 0.f};
#pragma unroll
  for (int k0 = 0; k0 < Cc; k0 += 32) {
    v16h a  = loadA_f16(xh + row0 * Cc + k0, Cc);
    v16h b0 = loadBt_f16(WqkvT + col0 * Cc + k0, Cc);
    v16h b1 = loadBt_f16(WqkvT + (col0 + 16) * Cc + k0, Cc);
    acc0 = wmma_f16(a, b0, acc0);
    acc1 = wmma_f16(a, b1, acc1);
  }

  int lane = threadIdx.x & 31;
  int n    = lane & 15;
  int mAdd = (lane >> 4) << 3;
  store_qkv_half(acc0, row0, col0,      n, mAdd, qbuf, kbuf, vT);
  store_qkv_half(acc1, row0, col0 + 16, n, mAdd, qbuf, kbuf, vT);
}

// ---------------------------------------------------------------------------
// Stage 2: flash attention with light-cone tile skipping.
// One wave per (batch, head, 16-query tile). Query tile = one (tq,hq) row,
// key tile = one (tk,hk) row, so t/h visibility is uniform per tile; only
// |wq-wk|<=r varies inside a tile.  Output written as f16 [BN][256].
// ---------------------------------------------------------------------------
__global__ __launch_bounds__(256) void lc_attn_kernel(
    const _Float16* __restrict__ qbuf, const _Float16* __restrict__ kbuf,
    const _Float16* __restrict__ vT, _Float16* __restrict__ Oh) {
  __shared__ __align__(32) _Float16 pscr[8][16 * 32];   // per-wave P tile

  int lane = threadIdx.x & 31;
  int wv   = threadIdx.x >> 5;
  int unit = blockIdx.x * 8 + wv;             // 1536 units
  int bat  = unit / (NH * QT);
  int head = (unit / QT) % NH;
  int qtile = unit % QT;
  int tq = qtile >> 4;
  int hq = qtile & 15;
  int q0 = qtile * 16;

  const float scale = 0.17677669529663687f;   // 1/sqrt(32)

  // Q operand: resident for the whole key loop.
  v16h qa = loadA_f16(qbuf + (bat * Nn + q0) * Cc + head * HD, Cc);

  v8f acc0 = {0.f, 0.f, 0.f, 0.f, 0.f, 0.f, 0.f, 0.f};  // out dims 0..15
  v8f acc1 = {0.f, 0.f, 0.f, 0.f, 0.f, 0.f, 0.f, 0.f};  // out dims 16..31
  float mrow[8], lrow[8];
#pragma unroll
  for (int e = 0; e < 8; ++e) { mrow[e] = -1e30f; lrow[e] = 0.f; }

  int n    = lane & 15;
  int mAdd = (lane >> 4) << 3;                // rows 0-7 / 8-15 per lane half

  const _Float16* kbase = kbuf + (bat * Nn) * Cc + head * HD;
  const _Float16* vbase = vT + (size_t)((bat * NH + head) * HD) * Nn;

  int tqm = tq > 1 ? tq : 1;
  int nch = (tq + 1) * 8;                     // causal: keys < (tq+1)*256
  for (int ch = 0; ch < nch; ++ch) {
    int kc0 = ch * 32;
    int tk  = ch >> 3;
    int r   = 1 + (15 * (tq - tk)) / tqm;     // floor, args non-negative
    int hk0 = (2 * ch) & 15;
    int hk1 = (2 * ch + 1) & 15;
    bool vis0 = iabs(hk0 - hq) <= r;
    bool vis1 = iabs(hk1 - hq) <= r;
    if (!vis0 && !vis1) continue;             // whole 32-key chunk masked

    // --- scores: S = Q K^T (per-lane contiguous loads from row-major K) ---
    v8f s0 = {0.f, 0.f, 0.f, 0.f, 0.f, 0.f, 0.f, 0.f};
    v8f s1 = {0.f, 0.f, 0.f, 0.f, 0.f, 0.f, 0.f, 0.f};
    if (vis0) {
      v16h bk = loadBt_f16(kbase + kc0 * Cc, Cc);
      v8f z = {0.f, 0.f, 0.f, 0.f, 0.f, 0.f, 0.f, 0.f};
      s0 = wmma_f16(qa, bk, z);
    }
    if (vis1) {
      v16h bk = loadBt_f16(kbase + (kc0 + 16) * Cc, Cc);
      v8f z = {0.f, 0.f, 0.f, 0.f, 0.f, 0.f, 0.f, 0.f};
      s1 = wmma_f16(qa, bk, z);
    }

    // --- online softmax; rows live per-VGPR split across lane halves -------
#pragma unroll
    for (int e = 0; e < 8; ++e) {
      int m = e + mAdd;
      bool wv0 = vis0 && (iabs(m - n) <= r);
      bool wv1 = vis1 && (iabs(m - n) <= r);
      float a = wv0 ? s0[e] * scale : -1e30f;
      float b = wv1 ? s1[e] * scale : -1e30f;
      float t = fmaxf(a, b);
#pragma unroll
      for (int off = 1; off < 16; off <<= 1)
        t = fmaxf(t, __shfl_xor(t, off, 32));
      float nm = fmaxf(mrow[e], t);
      float corr = (nm < -1e29f) ? 1.f : __expf(mrow[e] - nm);
      float p0 = (a < -1e29f) ? 0.f : __expf(a - nm);
      float p1 = (b < -1e29f) ? 0.f : __expf(b - nm);
      float ps = p0 + p1;
#pragma unroll
      for (int off = 1; off < 16; off <<= 1)
        ps += __shfl_xor(ps, off, 32);
      lrow[e] = lrow[e] * corr + ps;
      mrow[e] = nm;
      acc0[e] *= corr;
      acc1[e] *= corr;
      pscr[wv][m * 32 + n]      = (_Float16)p0;
      pscr[wv][m * 32 + 16 + n] = (_Float16)p1;
    }

    // --- O += P V : bounce P through LDS to re-layout D -> A operand -------
    v16h pa  = loadA_f16(&pscr[wv][0], 32);
    v16h bv0 = loadBt_f16(vbase + kc0, Nn);            // dims 0..15
    v16h bv1 = loadBt_f16(vbase + 16 * Nn + kc0, Nn);  // dims 16..31
    acc0 = wmma_f16(pa, bv0, acc0);
    acc1 = wmma_f16(pa, bv1, acc1);
  }

  // --- normalize and store f16 attention output [BN][256] ------------------
#pragma unroll
  for (int e = 0; e < 8; ++e) {
    int m = e + mAdd;
    float inv = 1.0f / lrow[e];
    int row = bat * Nn + q0 + m;
    Oh[row * Cc + head * HD + n]      = (_Float16)(acc0[e] * inv);
    Oh[row * Cc + head * HD + 16 + n] = (_Float16)(acc1[e] * inv);
  }
}

// ---------------------------------------------------------------------------
// Stage 3: out = O @ Wo + bo.  One wave per 16x32 tile of 3072x256.
// ---------------------------------------------------------------------------
__global__ __launch_bounds__(128) void lc_proj_kernel(
    const _Float16* __restrict__ Oh, const _Float16* __restrict__ WoT,
    const float* __restrict__ bo, float* __restrict__ out) {
  int wave = threadIdx.x >> 5;
  int tile = blockIdx.x * 4 + wave;           // 192 * 8 = 1536 tiles
  int rowT = tile >> 3;
  int colP = tile & 7;
  int row0 = rowT * 16;
  int col0 = colP * 32;

  v8f acc0 = {0.f, 0.f, 0.f, 0.f, 0.f, 0.f, 0.f, 0.f};
  v8f acc1 = {0.f, 0.f, 0.f, 0.f, 0.f, 0.f, 0.f, 0.f};
#pragma unroll
  for (int k0 = 0; k0 < Cc; k0 += 32) {
    v16h a  = loadA_f16(Oh + row0 * Cc + k0, Cc);
    v16h b0 = loadBt_f16(WoT + col0 * Cc + k0, Cc);
    v16h b1 = loadBt_f16(WoT + (col0 + 16) * Cc + k0, Cc);
    acc0 = wmma_f16(a, b0, acc0);
    acc1 = wmma_f16(a, b1, acc1);
  }

  int lane = threadIdx.x & 31;
  int n    = lane & 15;
  int mAdd = (lane >> 4) << 3;
  float bias0 = bo[col0 + n];
  float bias1 = bo[col0 + 16 + n];
#pragma unroll
  for (int e = 0; e < 8; ++e) {
    int m = e + mAdd;
    out[(row0 + m) * Cc + col0 + n]      = acc0[e] + bias0;
    out[(row0 + m) * Cc + col0 + 16 + n] = acc1[e] + bias1;
  }
}

// ---------------------------------------------------------------------------
// Launch. Workspace layout (~8.4 MB), S = BN*Cc*2 bytes = 1.5 MB:
//   [0)    xh    f16 [3072][256]
//   [1S)   Q     f16 [3072][256]
//   [2S)   K     f16 [3072][256]
//   [3S)   V^T   f16 [2][8][32][1536]
//   [4S)   Oh    f16 [3072][256]
//   [5S)   WqkvT f16 [768][256]   (384 KB)
//   [5S+384K) WoT f16 [256][256]  (128 KB)
// ---------------------------------------------------------------------------
extern "C" void kernel_launch(void* const* d_in, const int* in_sizes, int n_in,
                              void* d_out, int out_size, void* d_ws, size_t ws_size,
                              hipStream_t stream) {
  (void)in_sizes; (void)n_in; (void)out_size; (void)ws_size;
  const float* x    = (const float*)d_in[0];
  const float* Wqkv = (const float*)d_in[1];
  const float* Wo   = (const float*)d_in[2];
  const float* bo   = (const float*)d_in[3];
  float* out = (float*)d_out;

  const size_t S = (size_t)BN * Cc * 2;
  char* ws = (char*)d_ws;
  _Float16* xh    = (_Float16*)(ws);
  _Float16* qbuf  = (_Float16*)(ws + 1 * S);
  _Float16* kbuf  = (_Float16*)(ws + 2 * S);
  _Float16* vT    = (_Float16*)(ws + 3 * S);
  _Float16* Oh    = (_Float16*)(ws + 4 * S);
  _Float16* WqkvT = (_Float16*)(ws + 5 * S);
  _Float16* WoT   = (_Float16*)(ws + 5 * S + (size_t)WQN * 2);

  // 0) one-off convert/transpose: exactly 1M flat work-items
  lc_prep_kernel<<<(XN + WQN + WON) / 256, 256, 0, stream>>>(
      x, Wqkv, Wo, xh, WqkvT, WoT);
  // 1) 3072x768 in 16x32 tiles -> 4608 waves, 4 waves/block -> 1152 blocks
  lc_qkv_kernel<<<1152, 128, 0, stream>>>(xh, WqkvT, qbuf, kbuf, vT);
  // 2) 2*8*96 = 1536 wave-units, 8 waves/block -> 192 blocks
  lc_attn_kernel<<<192, 256, 0, stream>>>(qbuf, kbuf, vT, Oh);
  // 3) 3072x256 in 16x32 tiles -> 1536 waves, 4 waves/block -> 384 blocks
  lc_proj_kernel<<<384, 128, 0, stream>>>(Oh, WoT, bo, out);
}
